// NGE_56796647522784
// MI455X (gfx1250) — compile-verified, matrix operand order checked by reading, not run on hardware
//
#include <hip/hip_runtime.h>

typedef __attribute__((ext_vector_type(16))) _Float16 v16h;
typedef __attribute__((ext_vector_type(8)))  _Float16 v8h;
typedef __attribute__((ext_vector_type(8)))  float    v8f;
typedef __attribute__((ext_vector_type(4)))  float    v4f;

// ---------------------------------------------------------------------------
// Phase 1a: degree / dinv.  deg[j] = 1 (self) + sum of w over edges (i->j),
// edge (i,j) lives at flat index i*(i-1)/2 - 1 + j (complete-DAG layout).
// ---------------------------------------------------------------------------
__global__ void k_deg(const float* __restrict__ w, float* __restrict__ dinv, int n) {
  int j = blockIdx.x * blockDim.x + threadIdx.x;
  if (j >= n) return;
  float s = 1.0f;                       // self loop weight
  for (int i = 2; i < n; ++i) {
    if (i > j) {
      long long base = (long long)i * (i - 1) / 2 - 1;
      s += w[base + j];
    }
  }
  dinv[j] = rsqrtf(s);
}

// ---------------------------------------------------------------------------
// Phase 1b: dense h = x @ W  (tiny: <=2048x128x128 MACs)
// ---------------------------------------------------------------------------
__global__ void k_gemm(const float* __restrict__ x, const float* __restrict__ W,
                       float* __restrict__ h, int n, int din, int dout) {
  int idx = blockIdx.x * blockDim.x + threadIdx.x;
  if (idx >= n * dout) return;
  int c = idx % dout, r = idx / dout;
  const float* xr = x + (long long)r * din;
  float acc = 0.f;
  for (int k = 0; k < din; ++k) acc = fmaf(xr[k], W[k * dout + c], acc);
  h[idx] = acc;
}

// ---------------------------------------------------------------------------
// Phase 1c: suffix-scan aggregation for the complete DAG.
//   g[i] = dinv[i]*h[i];  out[j] = relu(dinv[j]*(S_excl(j) + g[j]) + bias)
//   S_excl(j) = sum_{i>j} g[i]   (j=0 special: no edge 1->0, subtract g[1])
// ---------------------------------------------------------------------------
__global__ void k_suf_partial(const float* __restrict__ h, const float* __restrict__ dinv,
                              float* __restrict__ P, int dout) {
  int t = blockIdx.x, c = threadIdx.x;
  float s = 0.f;
  int i0 = t * 128;
  for (int k = 0; k < 128; ++k) { int i = i0 + k; s += dinv[i] * h[(long long)i * dout + c]; }
  P[t * dout + c] = s;
}

__global__ void k_suf_base(const float* __restrict__ P, float* __restrict__ B,
                           int nt, int dout) {
  int c = threadIdx.x;
  float s = 0.f;
  for (int t = nt - 1; t >= 0; --t) { B[t * dout + c] = s; s += P[t * dout + c]; }
}

__global__ void k_suf_apply(const float* __restrict__ h, const float* __restrict__ dinv,
                            const float* __restrict__ B, const float* __restrict__ bias,
                            float* __restrict__ outf, _Float16* __restrict__ outh, int dout) {
  int t = blockIdx.x, c = threadIdx.x;
  float S = B[t * dout + c];                  // sum over all tiles above this one
  for (int k = 127; k >= 0; --k) {
    int i = t * 128 + k;
    float hv = h[(long long)i * dout + c];
    float di = dinv[i];
    float g  = di * hv;
    float Su = S;
    if (i == 0) Su = S - dinv[1] * h[dout + c];   // node 0 gets no edge from node 1
    float o = di * (Su + g) + bias[c];
    o = o > 0.f ? o : 0.f;                        // relu
    if (outf) outf[(long long)i * dout + c] = o;
    if (outh) outh[(long long)i * dout + c] = (_Float16)o;
    S += g;
  }
}

// ---------------------------------------------------------------------------
// Phase 1d: pre-format transposed weight matrices into WMMA A-operand blobs.
// A layout (16x32 f16): lane L: M = L&15; element t (=2v+s):
//   K = (v<4 ? 2v : 16+2(v-4)) + (L>=16 ? 8 : 0) + s
// Blob: fmt[op*512 + L*16 + t] -> each lane streams 32B coalesced per op.
// op index = kc*numMtiles + mt.
// ---------------------------------------------------------------------------
__device__ __forceinline__ int kpat(int t, int hi) {
  int v = t >> 1, s = t & 1;
  int b = (v < 4) ? 2 * v : 16 + 2 * (v - 4);
  return b + hi * 8 + s;
}

__global__ void k_fmt(const float* __restrict__ We1, const float* __restrict__ We2,
                      const float* __restrict__ Wop,
                      _Float16* __restrict__ A1, _Float16* __restrict__ A2,
                      _Float16* __restrict__ A3) {
  int b = blockIdx.x, tid = threadIdx.x;         // 26 blocks x 512 threads
  int L = tid >> 4, t = tid & 15;
  int hi = L >> 4, ml = L & 15;
  int kp = kpat(t, hi);
  if (b < 16) {            // layer1: A = We1^T, M=64 (4 mt), K=128 (4 kc)
    int kc = b >> 2, mt = b & 3;
    int k = kc * 32 + kp, m = mt * 16 + ml;
    A1[b * 512 + L * 16 + t] = (_Float16)We1[k * 64 + m];
  } else if (b < 24) {     // layer2: A = We2^T, M=64 (4 mt), K=64 (2 kc)
    int o = b - 16;
    int kc = o >> 2, mt = o & 3;
    int k = kc * 32 + kp, m = mt * 16 + ml;
    A2[o * 512 + L * 16 + t] = (_Float16)We2[k * 64 + m];
  } else {                 // layer3: A = Wop^T zero-padded to M=16, K=64 (2 kc)
    int o = b - 24;
    int k = o * 32 + kp, m = ml;
    A3[o * 512 + L * 16 + t] = (m < 8) ? (_Float16)Wop[k * 8 + m] : (_Float16)0.0f;
  }
}

// ---------------------------------------------------------------------------
// Phase 2: edge MLP.  One wave per PAIR of 16-edge tiles (32 edges) so each
// A-operand load feeds two WMMAs (halves weight-side L0 traffic per FLOP).
// Transposed orientation: D(out_feat x edges) = A(weights^T) x B(feat x edges).
// C/D layout (m = hi*8+r) matches B layout (k = hi*8+pat) -> each layer's
// accumulator becomes next layer's B operand in the SAME lane, no LDS.
// ---------------------------------------------------------------------------
union U16 { v16h v; v8h h2[2]; _Float16 h[16]; };

__global__ void __launch_bounds__(256) k_edge(
    const _Float16* __restrict__ hf, const int* __restrict__ edges,
    const _Float16* __restrict__ A1, const _Float16* __restrict__ A2,
    const _Float16* __restrict__ A3,
    const float* __restrict__ be1, const float* __restrict__ be2,
    const float* __restrict__ bop,
    float* __restrict__ out, long long E, int nnodes, int npairs) {
  int wave = threadIdx.x >> 5;
  int lane = threadIdx.x & 31;
  int pair = blockIdx.x * 8 + wave;                // tiles 2*pair, 2*pair+1
  if (pair >= npairs) return;                      // wave-uniform: EXEC stays full
  int nl = lane & 15, hi = lane >> 4;

  long long e[2]; int s[2], dn[2]; bool ok[2];
  const _Float16* srow[2]; const _Float16* drow[2];
  #pragma unroll
  for (int u = 0; u < 2; ++u) {
    long long eid = ((long long)pair * 2 + u) * 16 + nl;
    ok[u] = eid < E;
    e[u] = ok[u] ? eid : 0;
    s[u]  = edges[e[u]];
    dn[u] = edges[E + e[u]];
    srow[u] = hf + (long long)s[u]  * 64;
    drow[u] = hf + (long long)dn[u] * 64;
  }

  // ---- layer 1: K=128 (feat 0..63 = h[src], 64..127 = h[dst]), M=64 ----
  v8f d1[4][2];
  #pragma unroll
  for (int mt = 0; mt < 4; ++mt)
    #pragma unroll
    for (int u = 0; u < 2; ++u)
      d1[mt][u] = (v8f){0.f,0.f,0.f,0.f,0.f,0.f,0.f,0.f};

  #pragma unroll
  for (int kc = 0; kc < 4; ++kc) {
    U16 b[2];
    #pragma unroll
    for (int u = 0; u < 2; ++u) {
      const _Float16* rp = (kc < 2 ? srow[u] : drow[u]) + (kc & 1) * 32;
      b[u].h2[0] = *(const v8h*)(rp + hi * 8);        // dwords {0..3}+hi*4
      b[u].h2[1] = *(const v8h*)(rp + 16 + hi * 8);   // dwords {8..11}+hi*4
    }
    #pragma unroll
    for (int mt = 0; mt < 4; ++mt) {
      v16h a = *(const v16h*)(A1 + (kc * 4 + mt) * 512 + lane * 16);
      #pragma unroll
      for (int u = 0; u < 2; ++u)
        d1[mt][u] = __builtin_amdgcn_wmma_f32_16x16x32_f16(
            false, a, false, b[u].v, (short)0, d1[mt][u], false, false);
    }
  }

  // bias + relu, packed straight into layer-2 B operands (same-lane identity)
  U16 b2[2][2];                                      // [u][kc2]
  #pragma unroll
  for (int kc2 = 0; kc2 < 2; ++kc2) {
    v8f bvA = *(const v8f*)(be1 + (2 * kc2) * 16 + hi * 8);
    v8f bvB = *(const v8f*)(be1 + (2 * kc2 + 1) * 16 + hi * 8);
    #pragma unroll
    for (int u = 0; u < 2; ++u)
      #pragma unroll
      for (int t = 0; t < 8; ++t) {
        float x0 = d1[2 * kc2][u][t] + bvA[t];
        float x1 = d1[2 * kc2 + 1][u][t] + bvB[t];
        b2[u][kc2].h[t]     = (_Float16)(x0 > 0.f ? x0 : 0.f);
        b2[u][kc2].h[t + 8] = (_Float16)(x1 > 0.f ? x1 : 0.f);
      }
  }

  // ---- layer 2: K=64, M=64 ----
  v8f d2[4][2];
  #pragma unroll
  for (int mt = 0; mt < 4; ++mt)
    #pragma unroll
    for (int u = 0; u < 2; ++u)
      d2[mt][u] = (v8f){0.f,0.f,0.f,0.f,0.f,0.f,0.f,0.f};

  #pragma unroll
  for (int kc = 0; kc < 2; ++kc)
    #pragma unroll
    for (int mt = 0; mt < 4; ++mt) {
      v16h a = *(const v16h*)(A2 + (kc * 4 + mt) * 512 + lane * 16);
      #pragma unroll
      for (int u = 0; u < 2; ++u)
        d2[mt][u] = __builtin_amdgcn_wmma_f32_16x16x32_f16(
            false, a, false, b2[u][kc].v, (short)0, d2[mt][u], false, false);
    }

  U16 b3[2][2];
  #pragma unroll
  for (int kc2 = 0; kc2 < 2; ++kc2) {
    v8f bvA = *(const v8f*)(be2 + (2 * kc2) * 16 + hi * 8);
    v8f bvB = *(const v8f*)(be2 + (2 * kc2 + 1) * 16 + hi * 8);
    #pragma unroll
    for (int u = 0; u < 2; ++u)
      #pragma unroll
      for (int t = 0; t < 8; ++t) {
        float x0 = d2[2 * kc2][u][t] + bvA[t];
        float x1 = d2[2 * kc2 + 1][u][t] + bvB[t];
        b3[u][kc2].h[t]     = (_Float16)(x0 > 0.f ? x0 : 0.f);
        b3[u][kc2].h[t + 8] = (_Float16)(x1 > 0.f ? x1 : 0.f);
      }
  }

  // ---- layer 3: K=64, M=16 (8 real ops + 8 zero rows) ----
  v8f d3[2];
  #pragma unroll
  for (int u = 0; u < 2; ++u)
    d3[u] = (v8f){0.f,0.f,0.f,0.f,0.f,0.f,0.f,0.f};
  #pragma unroll
  for (int kc = 0; kc < 2; ++kc) {
    v16h a = *(const v16h*)(A3 + kc * 512 + lane * 16);
    #pragma unroll
    for (int u = 0; u < 2; ++u)
      d3[u] = __builtin_amdgcn_wmma_f32_16x16x32_f16(
          false, a, false, b3[u][kc].v, (short)0, d3[u], false, false);
  }

  // ---- softmax + mask: edge n's 8 logits sit entirely in lane n (hi==0) ----
  if (hi == 0) {
    #pragma unroll
    for (int u = 0; u < 2; ++u) {
      if (!ok[u]) continue;
      float lg[8];
      float mx = -3.4e38f;
      #pragma unroll
      for (int r = 0; r < 8; ++r) { lg[r] = d3[u][r] + bop[r]; mx = lg[r] > mx ? lg[r] : mx; }
      float se = 0.f;
      #pragma unroll
      for (int r = 0; r < 8; ++r) { lg[r] = __expf(lg[r] - mx); se += lg[r]; }
      float inv = 1.f / se;
      bool mk = s[u] < nnodes - 1;                   // forward loop skips src == n-1
      v4f o0, o1;
      #pragma unroll
      for (int r = 0; r < 4; ++r) {
        o0[r] = mk ? lg[r] * inv : 0.f;
        o1[r] = mk ? lg[4 + r] * inv : 0.f;
      }
      float* op = out + e[u] * 8;
      *(v4f*)op = o0;
      *((v4f*)op + 1) = o1;
    }
  }
}

// ---------------------------------------------------------------------------
extern "C" void kernel_launch(void* const* d_in, const int* in_sizes, int n_in,
                              void* d_out, int out_size, void* d_ws, size_t ws_size,
                              hipStream_t stream) {
  const float* embed = (const float*)d_in[0];
  const float* W1 = (const float*)d_in[1];  const float* b1 = (const float*)d_in[2];
  const float* W2 = (const float*)d_in[3];  const float* b2 = (const float*)d_in[4];
  const float* W3 = (const float*)d_in[5];  const float* b3 = (const float*)d_in[6];
  const float* We1 = (const float*)d_in[7]; const float* be1 = (const float*)d_in[8];
  const float* We2 = (const float*)d_in[9]; const float* be2 = (const float*)d_in[10];
  const float* Wop = (const float*)d_in[11]; const float* bop = (const float*)d_in[12];
  const int* edges = (const int*)d_in[13];
  const float* wts = (const float*)d_in[14];

  int n = in_sizes[0] / 64;                 // NUM_NODES (D=64)
  long long E = (long long)in_sizes[13] / 2;
  float* out = (float*)d_out;

  // scratch partition (256B aligned slabs)
  char* ws = (char*)d_ws;
  size_t off = 0;
  auto alloc = [&](size_t bytes) -> void* {
    void* p = ws + off;
    off += (bytes + 255) & ~(size_t)255;
    return p;
  };
  float*    dinv = (float*)alloc((size_t)n * 4);
  float*    hbuf = (float*)alloc((size_t)n * 128 * 4);
  float*    xbuf = (float*)alloc((size_t)n * 128 * 4);
  float*    P    = (float*)alloc(16 * 128 * 4);
  float*    Bb   = (float*)alloc(16 * 128 * 4);
  _Float16* hf   = (_Float16*)alloc((size_t)n * 64 * 2);
  _Float16* A1   = (_Float16*)alloc(16 * 512 * 2);
  _Float16* A2   = (_Float16*)alloc(8 * 512 * 2);
  _Float16* A3   = (_Float16*)alloc(2 * 512 * 2);

  int nt = n / 128;                          // 16 node tiles

  k_deg<<<(n + 255) / 256, 256, 0, stream>>>(wts, dinv, n);

  // GCN layer 1: 64 -> 128
  k_gemm<<<(n * 128 + 255) / 256, 256, 0, stream>>>(embed, W1, hbuf, n, 64, 128);
  k_suf_partial<<<nt, 128, 0, stream>>>(hbuf, dinv, P, 128);
  k_suf_base<<<1, 128, 0, stream>>>(P, Bb, nt, 128);
  k_suf_apply<<<nt, 128, 0, stream>>>(hbuf, dinv, Bb, b1, xbuf, nullptr, 128);

  // GCN layer 2: 128 -> 128
  k_gemm<<<(n * 128 + 255) / 256, 256, 0, stream>>>(xbuf, W2, hbuf, n, 128, 128);
  k_suf_partial<<<nt, 128, 0, stream>>>(hbuf, dinv, P, 128);
  k_suf_base<<<1, 128, 0, stream>>>(P, Bb, nt, 128);
  k_suf_apply<<<nt, 128, 0, stream>>>(hbuf, dinv, Bb, b2, xbuf, nullptr, 128);

  // GCN layer 3: 128 -> 64, emit f16 node features for the edge MLP
  k_gemm<<<(n * 64 + 255) / 256, 256, 0, stream>>>(xbuf, W3, hbuf, n, 128, 64);
  k_suf_partial<<<nt, 64, 0, stream>>>(hbuf, dinv, P, 64);
  k_suf_base<<<1, 64, 0, stream>>>(P, Bb, nt, 64);
  k_suf_apply<<<nt, 64, 0, stream>>>(hbuf, dinv, Bb, b3, nullptr, hf, 64);

  // Pre-format WMMA A-operands (weights^T) into per-lane blobs
  k_fmt<<<26, 512, 0, stream>>>(We1, We2, Wop, A1, A2, A3);

  // Edge MLP: one wave per pair of 16-edge tiles, 8 waves per block
  long long ntiles = (E + 15) / 16;
  int npairs = (int)((ntiles + 1) / 2);
  int blocks = (npairs + 7) / 8;
  k_edge<<<blocks, 256, 0, stream>>>(hf, edges, A1, A2, A3, be1, be2, bop,
                                     out, E, n, npairs);
}
